// MC_E_GCL_69088843923953
// MI455X (gfx1250) — matrix operand order repaired and checked by reading, not compiled, not run on hardware
//
#include <hip/hip_runtime.h>
#include <hip/hip_bf16.h>
#include <hip/hip_fp16.h>

#define N_NODES 20000
#define N_EDGES 320000
#define FDIM 128
#define WPB 4
#define NTILES ((N_NODES + 15) / 16)  // 1250

typedef __attribute__((ext_vector_type(16))) _Float16 v16h;
typedef __attribute__((ext_vector_type(8)))  _Float16 v8h;
typedef __attribute__((ext_vector_type(8)))  float    v8f;

union frag16 { v16h v; v8h h[2]; };

__device__ __forceinline__ v8f wmma_f16(v16h a, v16h b, v8f c) {
  // D = A(16x32 f16) * B(32x16 f16) + C(16x16 f32)
  return __builtin_amdgcn_wmma_f32_16x16x32_f16(false, a, false, b, (short)0, c,
                                                false, false);
}

__device__ __forceinline__ void wait_ds() {
  asm volatile("s_wait_dscnt 0" ::: "memory");
}

// A-fragment (16x32 f16) from row-major LDS tile; per ISA 7.12.2 the A K-map is
// lanes0-15: v0..3 = K0..7, v4..7 = K16..23 ; lanes16-31: +8.  Both halves are
// contiguous 8-half runs -> two 16B ds loads.
__device__ __forceinline__ v16h load_afrag(const _Float16* base, int ldm, int k0,
                                           int lane) {
  const int m = lane & 15, hf = lane >> 4;
  const _Float16* p = base + m * ldm + k0 + 8 * hf;
  frag16 u;
  u.h[0] = *(const v8h*)p;
  u.h[1] = *(const v8h*)(p + 16);
  return u.v;
}

// B-fragment from prepacked weights: block (kt,nt) is 512 halves, lane-major,
// 16 halves per lane -> two global_load_b128.
__device__ __forceinline__ v16h load_bfrag(const _Float16* w, int kt, int nt,
                                           int Nt, int lane) {
  const _Float16* p = w + (((kt * Nt + nt) << 9) + (lane << 4));
  frag16 u;
  u.h[0] = *(const v8h*)p;
  u.h[1] = *(const v8h*)(p + 8);
  return u.v;
}

// C/D layout: lane n = lane&15 is the column; vgpr j is row m = j + 8*(lane>>4).
// A bias is per-column -> splat one scalar across the 8 accumulator rows.
__device__ __forceinline__ v8f bias_init(const float* b, int n0, int nmax,
                                         int lane) {
  const int n = n0 + (lane & 15);
  const float v = (n < nmax) ? b[n] : 0.0f;
  v8f r = {v, v, v, v, v, v, v, v};
  return r;
}

__device__ __forceinline__ v8f silu8(v8f x) {
  v8f r;
#pragma unroll
  for (int j = 0; j < 8; ++j) r[j] = x[j] / (1.0f + __expf(-x[j]));
  return r;
}

__device__ __forceinline__ void store_c_f16(v8f acc, _Float16* dst, int ldd,
                                            int n0, int lane) {
  const int n = n0 + (lane & 15), hf = lane >> 4;
#pragma unroll
  for (int j = 0; j < 8; ++j) dst[(j + 8 * hf) * ldd + n] = (_Float16)acc[j];
}

// ---------------- weight prepack: fp32 row-major -> f16 B-fragment blocks ----
__global__ void prep_wfrag(const float* __restrict__ src, _Float16* __restrict__ dst,
                           int Ks, int Ns, int Nt, int total) {
  const int idx = blockIdx.x * blockDim.x + threadIdx.x;
  if (idx >= total) return;
  const int b = idx >> 9;
  const int w = idx & 511;
  const int lane = w >> 4;
  const int e = w & 15;
  const int kt = b / Nt, nt = b - kt * Nt;
  const int hf = lane >> 4, nl = lane & 15;
  const int j = e >> 1, pos = e & 1;
  const int k = kt * 32 + (e & 7) + 8 * hf + 16 * (e >> 3);  // B-matrix K map (ISA 7.12.2)
  const int n = nt * 16 + nl;
  const float v = (k < Ks && n < Ns) ? src[k * Ns + n] : 0.0f;
  dst[idx] = (_Float16)v;
}

__global__ void zero_f32(float* __restrict__ p, int n) {
  const int i = blockIdx.x * blockDim.x + threadIdx.x;
  if (i < n) p[i] = 0.0f;
}

// ---------------- fused edge pipeline: one wave per 16-edge tile -------------
__global__ __launch_bounds__(128) void edge_kernel(
    const float* __restrict__ h, const int* __restrict__ eidx,
    const float* __restrict__ coord,
    const float* __restrict__ eb1, const float* __restrict__ eb2,
    const float* __restrict__ cb1,
    const _Float16* __restrict__ We1, const _Float16* __restrict__ We2,
    const _Float16* __restrict__ Wc1, const _Float16* __restrict__ Wc2,
    float* __restrict__ agg_h, float* __restrict__ agg_c,
    float* __restrict__ cnt) {
  __shared__ _Float16 s_ein[WPB][16 * 288];  // edge_in tile (K padded 265->288)
  __shared__ _Float16 s_stg[WPB][16 * 32];   // hidden-pair staging
  __shared__ _Float16 s_ef [WPB][16 * 128];  // edge_feat (A for coord MLP)
  __shared__ float    s_cd [WPB][16 * 9];    // coord_diff
  __shared__ int      s_row[WPB][16];

  const int wv = threadIdx.x >> 5, lane = threadIdx.x & 31;
  const int tile = blockIdx.x * WPB + wv;    // grid is exact: 20000 tiles
  const int e0 = tile * 16;
  _Float16* ein = s_ein[wv];
  _Float16* stg = s_stg[wv];
  _Float16* ef  = s_ef[wv];

  // gather: coord_diff + radial (lanes 0..15, one edge each)
  if (lane < 16) {
    const int m = lane;
    const int r = eidx[e0 + m];
    const int c = eidx[N_EDGES + e0 + m];
    s_row[wv][m] = r;
    float cd[9];
#pragma unroll
    for (int q = 0; q < 9; ++q) {
      cd[q] = coord[r * 9 + q] - coord[c * 9 + q];
      s_cd[wv][m * 9 + q] = cd[q];
    }
#pragma unroll
    for (int i = 0; i < 3; ++i)
#pragma unroll
      for (int f = 0; f < 3; ++f) {
        const float s = cd[i * 3] * cd[f * 3] + cd[i * 3 + 1] * cd[f * 3 + 1] +
                        cd[i * 3 + 2] * cd[f * 3 + 2];
        ein[m * 288 + 256 + i * 3 + f] = (_Float16)s;
      }
#pragma unroll
    for (int q = 265; q < 288; ++q) ein[m * 288 + q] = (_Float16)0.0f;
  }
  // gather h[row], h[col]: feature-parallel across all 32 lanes (float4 loads)
  for (int m = 0; m < 16; ++m) {
    const int r = eidx[e0 + m];
    const int c = eidx[N_EDGES + e0 + m];
    const float4 hr = *(const float4*)(h + (long)r * FDIM + lane * 4);
    const float4 hc = *(const float4*)(h + (long)c * FDIM + lane * 4);
    _Float16* d0 = ein + m * 288 + lane * 4;
    d0[0] = (_Float16)hr.x; d0[1] = (_Float16)hr.y;
    d0[2] = (_Float16)hr.z; d0[3] = (_Float16)hr.w;
    _Float16* d1 = ein + m * 288 + 128 + lane * 4;
    d1[0] = (_Float16)hc.x; d1[1] = (_Float16)hc.y;
    d1[2] = (_Float16)hc.z; d1[3] = (_Float16)hc.w;
  }
  wait_ds();

  // edge MLP: (16x288)@(288x544) -> SiLU -> (16x544)@(544x128) -> SiLU
  v8f acc2[8];
#pragma unroll
  for (int nt = 0; nt < 8; ++nt) acc2[nt] = bias_init(eb2, nt * 16, 128, lane);

  for (int hp = 0; hp < 17; ++hp) {       // 544 hidden = 17 pairs of 16
    v8f h0 = bias_init(eb1, hp * 32,      530, lane);
    v8f h1 = bias_init(eb1, hp * 32 + 16, 530, lane);
#pragma unroll
    for (int k = 0; k < 9; ++k) {         // K = 288 = 9 * 32
      const v16h a = load_afrag(ein, 288, k * 32, lane);
      h0 = wmma_f16(a, load_bfrag(We1, k, hp * 2,     34, lane), h0);
      h1 = wmma_f16(a, load_bfrag(We1, k, hp * 2 + 1, 34, lane), h1);
    }
    h0 = silu8(h0); h1 = silu8(h1);
    store_c_f16(h0, stg, 32, 0, lane);
    store_c_f16(h1, stg, 32, 16, lane);
    wait_ds();
    const v16h a2 = load_afrag(stg, 32, 0, lane);
#pragma unroll
    for (int nt = 0; nt < 8; ++nt)
      acc2[nt] = wmma_f16(a2, load_bfrag(We2, hp, nt, 8, lane), acc2[nt]);
    wait_ds();
  }
#pragma unroll
  for (int nt = 0; nt < 8; ++nt) acc2[nt] = silu8(acc2[nt]);  // edge_feat (f32)

  // stash edge_feat as f16 A-tile for the coord MLP
#pragma unroll
  for (int nt = 0; nt < 8; ++nt) store_c_f16(acc2[nt], ef, 128, nt * 16, lane);
  wait_ds();

  // coord MLP: (16x128)@(128x256) -> SiLU -> (16x256)@(256x16, cols 3..15 = 0)
  v8f phi = {0, 0, 0, 0, 0, 0, 0, 0};
  for (int hp = 0; hp < 8; ++hp) {
    v8f c0 = bias_init(cb1, hp * 32,      256, lane);
    v8f c1 = bias_init(cb1, hp * 32 + 16, 256, lane);
#pragma unroll
    for (int k = 0; k < 4; ++k) {
      const v16h a = load_afrag(ef, 128, k * 32, lane);
      c0 = wmma_f16(a, load_bfrag(Wc1, k, hp * 2,     16, lane), c0);
      c1 = wmma_f16(a, load_bfrag(Wc1, k, hp * 2 + 1, 16, lane), c1);
    }
    c0 = silu8(c0); c1 = silu8(c1);
    store_c_f16(c0, stg, 32, 0, lane);
    store_c_f16(c1, stg, 32, 16, lane);
    wait_ds();
    const v16h a2 = load_afrag(stg, 32, 0, lane);
    phi = wmma_f16(a2, load_bfrag(Wc2, hp, 0, 1, lane), phi);
    wait_ds();
  }

  // scatter: agg_h += edge_feat ; agg_c += coord_diff * phi ; cnt += 1
  const int hf = lane >> 4, nl = lane & 15;
#pragma unroll
  for (int nt = 0; nt < 8; ++nt)
#pragma unroll
    for (int j = 0; j < 8; ++j) {
      const int m = j + 8 * hf;
      const int r = s_row[wv][m];
      __hip_atomic_fetch_add(&agg_h[(long)r * FDIM + nt * 16 + nl], acc2[nt][j],
                             __ATOMIC_RELAXED, __HIP_MEMORY_SCOPE_AGENT);
    }
  if (nl < 3) {
#pragma unroll
    for (int j = 0; j < 8; ++j) {
      const int m = j + 8 * hf;
      const int r = s_row[wv][m];
      const float p = phi[j];
#pragma unroll
      for (int d = 0; d < 3; ++d) {
        const float t = s_cd[wv][m * 9 + nl * 3 + d] * p;
        __hip_atomic_fetch_add(&agg_c[(long)r * 9 + nl * 3 + d], t,
                               __ATOMIC_RELAXED, __HIP_MEMORY_SCOPE_AGENT);
      }
    }
  }
  if (lane < 16)
    __hip_atomic_fetch_add(&cnt[s_row[wv][lane]], 1.0f, __ATOMIC_RELAXED,
                           __HIP_MEMORY_SCOPE_AGENT);
}

// ---------------- node MLP + coord epilogue: one wave per 16-node tile -------
__global__ __launch_bounds__(128) void node_kernel(
    const float* __restrict__ h, const float* __restrict__ coord,
    const float* __restrict__ nb1, const float* __restrict__ nb2,
    const _Float16* __restrict__ Wn1, const _Float16* __restrict__ Wn2,
    const float* __restrict__ agg_h, const float* __restrict__ agg_c,
    const float* __restrict__ cnt,
    float* __restrict__ out_h, float* __restrict__ out_c) {
  __shared__ _Float16 s_in [WPB][16 * 256];
  __shared__ _Float16 s_stg[WPB][16 * 32];
  const int wv = threadIdx.x >> 5, lane = threadIdx.x & 31;
  const int tile = blockIdx.x * WPB + wv;
  if (tile >= NTILES) return;  // wave-uniform: EXEC stays all-ones for WMMA
  const int n0 = tile * 16;
  _Float16* nin = s_in[wv];
  _Float16* stg = s_stg[wv];

  for (int m = 0; m < 16; ++m) {
    const float4 hv = *(const float4*)(h     + (long)(n0 + m) * FDIM + lane * 4);
    const float4 av = *(const float4*)(agg_h + (long)(n0 + m) * FDIM + lane * 4);
    _Float16* d0 = nin + m * 256 + lane * 4;
    d0[0] = (_Float16)hv.x; d0[1] = (_Float16)hv.y;
    d0[2] = (_Float16)hv.z; d0[3] = (_Float16)hv.w;
    _Float16* d1 = nin + m * 256 + 128 + lane * 4;
    d1[0] = (_Float16)av.x; d1[1] = (_Float16)av.y;
    d1[2] = (_Float16)av.z; d1[3] = (_Float16)av.w;
  }
  wait_ds();

  v8f acc[8];
#pragma unroll
  for (int nt = 0; nt < 8; ++nt) acc[nt] = bias_init(nb2, nt * 16, 128, lane);

  for (int hp = 0; hp < 16; ++hp) {       // 512 hidden = 16 pairs of 16
    v8f h0 = bias_init(nb1, hp * 32,      512, lane);
    v8f h1 = bias_init(nb1, hp * 32 + 16, 512, lane);
#pragma unroll
    for (int k = 0; k < 8; ++k) {         // K = 256
      const v16h a = load_afrag(nin, 256, k * 32, lane);
      h0 = wmma_f16(a, load_bfrag(Wn1, k, hp * 2,     32, lane), h0);
      h1 = wmma_f16(a, load_bfrag(Wn1, k, hp * 2 + 1, 32, lane), h1);
    }
    h0 = silu8(h0); h1 = silu8(h1);
    store_c_f16(h0, stg, 32, 0, lane);
    store_c_f16(h1, stg, 32, 16, lane);
    wait_ds();
    const v16h a2 = load_afrag(stg, 32, 0, lane);
#pragma unroll
    for (int nt = 0; nt < 8; ++nt)
      acc[nt] = wmma_f16(a2, load_bfrag(Wn2, hp, nt, 8, lane), acc[nt]);
    wait_ds();
  }

  const int hf = lane >> 4, nl = lane & 15;
#pragma unroll
  for (int nt = 0; nt < 8; ++nt) {
    const v8f r = silu8(acc[nt]);
#pragma unroll
    for (int j = 0; j < 8; ++j) {
      const int m = j + 8 * hf;
      const long idx = (long)(n0 + m) * FDIM + nt * 16 + nl;
      out_h[idx] = h[idx] + r[j];  // residual
    }
  }
  for (int q = lane; q < 144; q += 32) {  // 16 nodes x 9 coord values
    const int m = q / 9, cdx = q - m * 9;
    const long ni = n0 + m;
    float c = cnt[ni];
    c = fmaxf(c, 1.0f);
    float a = agg_c[ni * 9 + cdx] / c;
    a = fminf(fmaxf(a, -10.0f), 10.0f);
    out_c[ni * 9 + cdx] = coord[ni * 9 + cdx] + a;
  }
}

extern "C" void kernel_launch(void* const* d_in, const int* in_sizes, int n_in,
                              void* d_out, int out_size, void* d_ws,
                              size_t ws_size, hipStream_t stream) {
  const float* h     = (const float*)d_in[0];
  const int*   eidx  = (const int*)d_in[1];
  const float* coord = (const float*)d_in[2];
  const float* eW1 = (const float*)d_in[3];
  const float* eb1 = (const float*)d_in[4];
  const float* eW2 = (const float*)d_in[5];
  const float* eb2 = (const float*)d_in[6];
  const float* nW1 = (const float*)d_in[7];
  const float* nb1 = (const float*)d_in[8];
  const float* nW2 = (const float*)d_in[9];
  const float* nb2 = (const float*)d_in[10];
  const float* cW1 = (const float*)d_in[11];
  const float* cb1 = (const float*)d_in[12];
  const float* cW2 = (const float*)d_in[13];

  char* ws = (char*)d_ws;
  _Float16* We1 = (_Float16*)(ws + 0);        // 288x544 -> 313344 B
  _Float16* We2 = (_Float16*)(ws + 313344);   // 544x128 -> 139264 B
  _Float16* Wn1 = (_Float16*)(ws + 452608);   // 256x512 -> 262144 B
  _Float16* Wn2 = (_Float16*)(ws + 714752);   // 512x128 -> 131072 B
  _Float16* Wc1 = (_Float16*)(ws + 845824);   // 128x256 ->  65536 B
  _Float16* Wc2 = (_Float16*)(ws + 911360);   // 256x16  ->   8192 B
  float* agg_h = (float*)(ws + 1048576);                       // N*128 f32
  float* agg_c = (float*)(ws + 1048576 + 10240000);            // N*9 f32
  float* cnt   = (float*)(ws + 1048576 + 10240000 + 720000);   // N f32

  {  // zero the (contiguous) accumulator region every call
    const int n = N_NODES * FDIM + N_NODES * 9 + N_NODES;
    zero_f32<<<(n + 255) / 256, 256, 0, stream>>>(agg_h, n);
  }
  auto prep = [&](const float* src, _Float16* dst, int Ks, int Ns, int Kt,
                  int Nt) {
    const int total = Kt * Nt * 512;
    prep_wfrag<<<(total + 255) / 256, 256, 0, stream>>>(src, dst, Ks, Ns, Nt,
                                                        total);
  };
  prep(eW1, We1, 265, 530,  9, 34);
  prep(eW2, We2, 530, 128, 17,  8);
  prep(nW1, Wn1, 256, 512,  8, 32);
  prep(nW2, Wn2, 512, 128, 16,  8);
  prep(cW1, Wc1, 128, 256,  4, 16);
  prep(cW2, Wc2, 256,   3,  8,  1);

  edge_kernel<<<N_EDGES / (16 * WPB), 128, 0, stream>>>(
      h, eidx, coord, eb1, eb2, cb1, We1, We2, Wc1, Wc2, agg_h, agg_c, cnt);

  float* out_h = (float*)d_out;
  float* out_c = out_h + (long)N_NODES * FDIM;
  node_kernel<<<(NTILES + WPB - 1) / WPB, 128, 0, stream>>>(
      h, coord, nb1, nb2, Wn1, Wn2, agg_h, agg_c, cnt, out_h, out_c);
}
